// SparseAttention_80367428043283
// MI455X (gfx1250) — matrix operand order repaired and checked by reading, not compiled, not run on hardware
//
#include <hip/hip_runtime.h>

typedef __attribute__((ext_vector_type(16))) __bf16 v16bf;
typedef __attribute__((ext_vector_type(8)))  float  v8f;

namespace {
constexpr int kB  = 16;
constexpr int kT  = 2048;
constexpr int kD  = 256;
constexpr int kQT = 16;                    // query rows per workgroup
constexpr int kNW = 8;                     // waves per workgroup
constexpr int kThreads = kNW * 32;
constexpr int kQTiles = kT / kQT;          // 128
constexpr int kColsPerWave = kT / kNW;     // 256
constexpr int kSP = kT + 4;                // padded LDS row stride (bank-conflict free)
constexpr float kAlphaM1 = 0.5f;           // alpha - 1
}

__global__ __launch_bounds__(kThreads) void entmax_attn_kernel(
    const float* __restrict__ Q, const float* __restrict__ V,
    const float* __restrict__ K, float* __restrict__ O) {
  __shared__ float S[kQT][kSP];            // scores, then normalized probabilities

  const int lane = threadIdx.x & 31;
  const int wave = threadIdx.x >> 5;
  const int l15  = lane & 15;
  const int hi16 = lane >> 4;

  const int b     = blockIdx.x / kQTiles;
  const int qbase = (blockIdx.x % kQTiles) * kQT;

  const float* __restrict__ Qb = Q + ((size_t)b * kT + qbase) * kD;
  const float* __restrict__ Kb = K + (size_t)b * kT * kD;
  const float* __restrict__ Vb = V + (size_t)b * kT * kD;
  float* __restrict__ Ob       = O + ((size_t)b * kT + qbase) * kD;

  // ---------------- Phase 1: S = Q . K^T  (bf16 WMMA, fp32 accum) ----------
  v16bf aq[kD / 32];                        // Q A-fragments, reused for all key tiles
  {
    const float* qrow = Qb + (size_t)l15 * kD;
    const int off = hi16 * 16;              // A 16x32 in this code's k order: contiguous K16 per half-wave (same as B)
#pragma unroll
    for (int j = 0; j < kD / 32; ++j) {
      v16bf a;
#pragma unroll
      for (int i = 0; i < 16; ++i) {
        const int k = j * 32 + off + i;
        a[i] = (__bf16)qrow[k];
      }
      aq[j] = a;
    }
  }

  for (int ct = 0; ct < kColsPerWave / 16; ++ct) {
    const int keybase = wave * kColsPerWave + ct * 16;
    const float* krow = Kb + (size_t)(keybase + l15) * kD;   // lane = B column = key row
    v8f acc = {};
#pragma unroll
    for (int j = 0; j < kD / 32; ++j) {
      const int kb0 = j * 32 + hi16 * 16;   // B 32x16 in this code's k order: contiguous K16 per half-wave (same as A)
      v16bf bk;
#pragma unroll
      for (int i = 0; i < 16; ++i) bk[i] = (__bf16)krow[kb0 + i];
      acc = __builtin_amdgcn_wmma_f32_16x16x32_bf16(false, aq[j], false, bk,
                                                    (short)0, acc, false, false);
    }
#pragma unroll
    for (int r = 0; r < 8; ++r)             // C/D layout: M = 8*hi16 + r, N = l15
      S[hi16 * 8 + r][keybase + l15] = acc[r];
  }
  __syncthreads();

  // ---------------- Phase 2: alpha-entmax bisection (alpha=1.5 -> square) --
  {
    const int r0 = wave * 2, r1 = r0 + 1;   // each wave owns two rows
    float z0[kT / 32], z1[kT / 32];         // register-cached z = (alpha-1)*score
#pragma unroll
    for (int c = 0; c < kT / 32; ++c) {
      z0[c] = kAlphaM1 * S[r0][lane + c * 32];
      z1[c] = kAlphaM1 * S[r1][lane + c * 32];
    }
    float m0 = -3.0e38f, m1 = -3.0e38f;
#pragma unroll
    for (int c = 0; c < kT / 32; ++c) {
      m0 = fmaxf(m0, z0[c]);
      m1 = fmaxf(m1, z1[c]);
    }
#pragma unroll
    for (int s = 16; s > 0; s >>= 1) {      // wave32 butterfly: all lanes get row max
      m0 = fmaxf(m0, __shfl_xor(m0, s, 32));
      m1 = fmaxf(m1, __shfl_xor(m1, s, 32));
    }
    float lo0 = m0 - 1.0f, hi0 = m0;
    float lo1 = m1 - 1.0f, hi1 = m1;
    for (int it = 0; it < 50; ++it) {
      const float t0 = 0.5f * (lo0 + hi0);
      const float t1 = 0.5f * (lo1 + hi1);
      float f0 = 0.0f, f1 = 0.0f;
#pragma unroll
      for (int c = 0; c < kT / 32; ++c) {
        const float a0 = fmaxf(z0[c] - t0, 0.0f);
        const float a1 = fmaxf(z1[c] - t1, 0.0f);
        f0 = fmaf(a0, a0, f0);              // q = 1/(alpha-1) = 2 -> square
        f1 = fmaf(a1, a1, f1);
      }
#pragma unroll
      for (int s = 16; s > 0; s >>= 1) {
        f0 += __shfl_xor(f0, s, 32);
        f1 += __shfl_xor(f1, s, 32);
      }
      if (f0 - 1.0f > 0.0f) lo0 = t0; else hi0 = t0;
      if (f1 - 1.0f > 0.0f) lo1 = t1; else hi1 = t1;
    }
    const float t0 = 0.5f * (lo0 + hi0);
    const float t1 = 0.5f * (lo1 + hi1);
    float s0 = 0.0f, s1 = 0.0f;
#pragma unroll
    for (int c = 0; c < kT / 32; ++c) {
      float a0 = fmaxf(z0[c] - t0, 0.0f); a0 *= a0; z0[c] = a0; s0 += a0;
      float a1 = fmaxf(z1[c] - t1, 0.0f); a1 *= a1; z1[c] = a1; s1 += a1;
    }
#pragma unroll
    for (int s = 16; s > 0; s >>= 1) {
      s0 += __shfl_xor(s0, s, 32);
      s1 += __shfl_xor(s1, s, 32);
    }
    const float i0 = 1.0f / s0, i1 = 1.0f / s1;
#pragma unroll
    for (int c = 0; c < kT / 32; ++c) {     // exact renormalization, write back in place
      S[r0][lane + c * 32] = z0[c] * i0;
      S[r1][lane + c * 32] = z1[c] * i1;
    }
  }
  __syncthreads();

  // ---------------- Phase 3: O = P . V  (bf16 WMMA, fp32 accum) ------------
#pragma unroll
  for (int t = 0; t < 2; ++t) {
    const int colbase = wave * 32 + t * 16;
    v8f acc = {};
    for (int kb = 0; kb < kT / 32; ++kb) {
      v16bf ap;                             // A = probability rows from LDS
      const int off = hi16 * 16;
#pragma unroll
      for (int i = 0; i < 16; ++i) {
        const int k = kb * 32 + off + i;
        ap[i] = (__bf16)S[l15][k];
      }
      v16bf bv;                             // B = V tile: lane = output dim column
      const int kr = kb * 32 + hi16 * 16;
#pragma unroll
      for (int i = 0; i < 16; ++i)
        bv[i] = (__bf16)Vb[(size_t)(kr + i) * kD + colbase + l15];
      acc = __builtin_amdgcn_wmma_f32_16x16x32_bf16(false, ap, false, bv,
                                                    (short)0, acc, false, false);
    }
#pragma unroll
    for (int r = 0; r < 8; ++r)
      Ob[(size_t)(hi16 * 8 + r) * kD + colbase + l15] = acc[r];
  }
}

extern "C" void kernel_launch(void* const* d_in, const int* in_sizes, int n_in,
                              void* d_out, int out_size, void* d_ws, size_t ws_size,
                              hipStream_t stream) {
  // setup_inputs() dict order: query, value, key (all fp32)
  const float* Q = (const float*)d_in[0];
  const float* V = (const float*)d_in[1];
  const float* K = (const float*)d_in[2];
  float* O = (float*)d_out;
  (void)in_sizes; (void)n_in; (void)out_size; (void)d_ws; (void)ws_size;

  dim3 grid(kB * kQTiles);    // 16 batches * 128 query tiles = 2048 workgroups
  dim3 block(kThreads);       // 8 waves (wave32)
  hipLaunchKernelGGL(entmax_attn_kernel, grid, block, 0, stream, Q, V, K, O);
}
